// Attention_RoPE_47639777247805
// MI455X (gfx1250) — matrix operand order repaired and checked by reading, not compiled, not run on hardware
//
#include <hip/hip_runtime.h>

typedef __bf16 bf16;
typedef __attribute__((ext_vector_type(16))) __bf16 v16bf;
typedef __attribute__((ext_vector_type(8)))  float   v8f;
typedef __attribute__((ext_vector_type(4)))  unsigned int u32x4;
typedef __attribute__((ext_vector_type(8)))  int i32x8;
typedef __attribute__((ext_vector_type(4)))  int i32x4;

#define DIM       1024
#define HEADS     16
#define DIM_HEAD  64
#define SEQ       2048
#define BATCH     4
#define ROWS      (BATCH*SEQ)   /* 8192 */
#define QKV_N     (3*DIM)       /* 3072 */

union FragAB {
    v16bf bf;
    u32x4 q[2];
};

__device__ __forceinline__ v8f wmma_bf16(const FragAB& a, const FragAB& b, v8f c) {
    return __builtin_amdgcn_wmma_f32_16x16x32_bf16(
        /*neg_a=*/false, a.bf, /*neg_b=*/false, b.bf,
        /*c_mod=*/(short)0, c, /*reuse_a=*/false, /*reuse_b=*/false);
}

// ---- Tensor Data Mover: 2D tile (bf16) global -> LDS, D# per ISA 8.3/8.4 ----
// tile_d0 = contiguous elements per line, tile_d1 = lines, stride = elements
// between lines. LDS receives tile_d0*tile_d1 contiguous bf16 at lds_off.
// Trailing empty asm escapes the LDS destination with a memory clobber so the
// compiler knows the TDM wrote that memory.
__device__ __forceinline__ void tdm_load_2d(bf16* lds_dst, unsigned lds_off,
                                            const bf16* gptr,
                                            unsigned tile_d0, unsigned tile_d1,
                                            unsigned stride,
                                            unsigned tensor_d0, unsigned tensor_d1) {
    unsigned long long ga = (unsigned long long)(uintptr_t)gptr;
    u32x4 g0;
    g0[0] = 1u;                                     // count=1, user mode
    g0[1] = lds_off;                                // lds_addr (bytes)
    g0[2] = (unsigned)ga;                           // global_addr[31:0]
    g0[3] = (unsigned)(ga >> 32) | (2u << 30);      // global_addr[56:32] | type=2
    i32x8 g1;
    g1[0] = (int)(1u << 16);                        // wg_mask=0, data_size=1 (2B)
    g1[1] = (int)((tensor_d0 & 0xFFFFu) << 16);     // tensor_dim0[15:0]
    g1[2] = (int)((tensor_d0 >> 16) | ((tensor_d1 & 0xFFFFu) << 16));
    g1[3] = (int)((tensor_d1 >> 16) | (tile_d0 << 16));
    g1[4] = (int)tile_d1;                           // tile_dim1, tile_dim2=0
    g1[5] = (int)stride;                            // tensor_dim0_stride[31:0]
    g1[6] = 0;                                      // stride[47:32]=0, dim1_stride=0
    g1[7] = 0;
    i32x4 gz4;
    gz4[0] = 0; gz4[1] = 0; gz4[2] = 0; gz4[3] = 0;
    i32x8 gz8;
    gz8[0] = 0; gz8[1] = 0; gz8[2] = 0; gz8[3] = 0;
    gz8[4] = 0; gz8[5] = 0; gz8[6] = 0; gz8[7] = 0;
    __builtin_amdgcn_tensor_load_to_lds(g0, g1, gz4, gz4, gz8, 0);
    asm volatile("" : : "r"(lds_dst) : "memory");   // TDM writes through lds_dst
}

// ---------------- Kernel 1: weight pack fp32[K][N] -> bf16[N][K] ----------------
__global__ void pack_w_kernel(const float* __restrict__ w, bf16* __restrict__ wp, int K, int N) {
    int idx = blockIdx.x * 256 + threadIdx.x;
    int n = idx / K;
    int k = idx - n * K;
    if (n < N) wp[(size_t)n * K + k] = (bf16)w[(size_t)k * N + n];
}

// ---------------- Kernel 2: LayerNorm fp32 -> bf16 ----------------
__global__ void ln_kernel(const float* __restrict__ x, const float* __restrict__ g,
                          const float* __restrict__ be, bf16* __restrict__ xn) {
    __shared__ float s1[256];
    __shared__ float s2[256];
    int row = blockIdx.x;
    const float4* xr = (const float4*)(x + (size_t)row * DIM);
    float4 v = xr[threadIdx.x];
    s1[threadIdx.x] = v.x + v.y + v.z + v.w;
    s2[threadIdx.x] = v.x * v.x + v.y * v.y + v.z * v.z + v.w * v.w;
    __syncthreads();
    for (int off = 128; off > 0; off >>= 1) {
        if (threadIdx.x < off) {
            s1[threadIdx.x] += s1[threadIdx.x + off];
            s2[threadIdx.x] += s2[threadIdx.x + off];
        }
        __syncthreads();
    }
    float mu  = s1[0] * (1.0f / DIM);
    float var = s2[0] * (1.0f / DIM) - mu * mu;
    float inv = rsqrtf(var + 1e-5f);
    int c = threadIdx.x * 4;
    bf16* o = xn + (size_t)row * DIM + c;
    o[0] = (bf16)((v.x - mu) * inv * g[c + 0] + be[c + 0]);
    o[1] = (bf16)((v.y - mu) * inv * g[c + 1] + be[c + 1]);
    o[2] = (bf16)((v.z - mu) * inv * g[c + 2] + be[c + 2]);
    o[3] = (bf16)((v.w - mu) * inv * g[c + 3] + be[c + 3]);
}

// ---- Shared GEMM mainloop: TDM double-buffered LDS, block tile 64M x 128N ----
// smem element layout (bf16): A0 @0 (2048), A1 @2048, B0 @4096 (4096), B1 @8192.
// LDS byte offsets (module LDS base): A0=0, A1=4096, B0=8192, B1=16384.
__device__ __forceinline__ void gemm_mainloop(const bf16* __restrict__ A,
                                              const bf16* __restrict__ Bp,
                                              int m0blk, int n0blk, int KDIM, int rowsA,
                                              bf16* smem, int lane, int wv, int wm, int wn,
                                              v8f acc[4]) {
    if (wv == 0) {
        tdm_load_2d(smem,        0,    A  + (size_t)m0blk * KDIM, 32, 64,  KDIM, KDIM, rowsA);
        tdm_load_2d(smem + 4096, 8192, Bp + (size_t)n0blk * KDIM, 32, 128, KDIM, KDIM, QKV_N);
        __builtin_amdgcn_s_wait_tensorcnt(0);
    }
    __syncthreads();
    for (int kk = 0; kk < KDIM; kk += 32) {
        int cur = (kk >> 5) & 1;
        if (wv == 0 && (kk + 32) < KDIM) {
            tdm_load_2d(smem + (cur ? 0 : 2048), cur ? 0 : 4096,
                        A + (size_t)m0blk * KDIM + kk + 32, 32, 64, KDIM, KDIM, rowsA);
            tdm_load_2d(smem + (cur ? 4096 : 8192), cur ? 8192 : 16384,
                        Bp + (size_t)n0blk * KDIM + kk + 32, 32, 128, KDIM, KDIM, QKV_N);
        }
        const bf16* sA = smem + cur * 2048;
        const bf16* sB = smem + 4096 + cur * 4096;
        // Batch all fragment loads (1xA + 4xB = 10 ds_load_b128) before WMMAs
        // so the scheduler can overlap LDS returns with matrix issue.
        FragAB a, bfr[4];
        {
            const bf16* p = sA + (wm * 16 + (lane & 15)) * 32 + ((lane >> 4) << 3);
            a.q[0] = *(const u32x4*)p;
            a.q[1] = *(const u32x4*)(p + 16);
        }
#pragma unroll
        for (int t = 0; t < 4; t++) {
            const bf16* p = sB + (wn * 64 + t * 16 + (lane & 15)) * 32 + ((lane >> 4) << 3);
            bfr[t].q[0] = *(const u32x4*)p;
            bfr[t].q[1] = *(const u32x4*)(p + 16);
        }
#pragma unroll
        for (int t = 0; t < 4; t++)
            acc[t] = wmma_bf16(a, bfr[t], acc[t]);
        if (wv == 0) __builtin_amdgcn_s_wait_tensorcnt(0);
        __syncthreads();
    }
}

// ---------------- Kernel 3: QKV GEMM + RoPE + head scatter ----------------
__global__ void __launch_bounds__(256, 1)
qkv_gemm_rope(const bf16* __restrict__ xn, const bf16* __restrict__ wp,
              bf16* __restrict__ qb, bf16* __restrict__ kb,
              bf16* __restrict__ vT) {
    __shared__ __align__(16) bf16 smem[12288];      // 24 KB
    int lane = threadIdx.x & 31;
    int wv   = threadIdx.x >> 5;
    int wm = wv & 3, wn = wv >> 2;
    int m0 = blockIdx.y * 64 + wm * 16;
    int n0 = blockIdx.x * 128 + wn * 64;
    v8f acc[4] = {};
    gemm_mainloop(xn, wp, blockIdx.y * 64, blockIdx.x * 128, DIM, ROWS,
                  smem, lane, wv, wm, wn, acc);
    int half = lane >> 4;
    if (n0 < 2 * DIM) {
        bf16* dst = (n0 < DIM) ? qb : kb;
        int h = ((n0 < DIM) ? n0 : (n0 - DIM)) / DIM_HEAD;
#pragma unroll
        for (int t = 0; t < 2; t++) {
            int j = t * 16 + (lane & 15);                 // [0,32)
            float invf = __expf(-(float)j * (9.210340372f / 32.0f)); // 1/10000^(j/32)
#pragma unroll
            for (int r = 0; r < 8; r++) {
                int gr = m0 + r + 8 * half;
                int bi = gr >> 11, s = gr & (SEQ - 1);
                float ang = (float)s * invf;
                float cs = __cosf(ang), sn = __sinf(ang);
                float x1 = acc[t][r], x2 = acc[t + 2][r];
                size_t base = (((size_t)bi * HEADS + h) * SEQ + s) * DIM_HEAD;
                dst[base + j]      = (bf16)(x1 * cs - x2 * sn);
                dst[base + j + 32] = (bf16)(x1 * sn + x2 * cs);
            }
        }
    } else {
        int h = (n0 - 2 * DIM) / DIM_HEAD;
#pragma unroll
        for (int t = 0; t < 4; t++) {
            int j = t * 16 + (lane & 15);                 // [0,64)
#pragma unroll
            for (int r = 0; r < 8; r++) {
                int gr = m0 + r + 8 * half;
                int bi = gr >> 11, s = gr & (SEQ - 1);
                vT[(((size_t)bi * HEADS + h) * DIM_HEAD + j) * SEQ + s] = (bf16)acc[t][r];
            }
        }
    }
}

// ---------------- Kernel 4: flash attention per (b,h) ----------------
// smem (bf16 elems): K0 @0, K1 @2048, V0 @4096, V1 @6144, P @8192 + wv*512.
// LDS byte offsets: K0=0, K1=4096, V0=8192, V1=12288, P=16384.
__global__ void __launch_bounds__(256, 1)
attn_kernel(const bf16* __restrict__ qb, const bf16* __restrict__ kb,
            const bf16* __restrict__ vT, bf16* __restrict__ ob) {
    __shared__ __align__(16) bf16 smem[12288];      // 24 KB
    int lane = threadIdx.x & 31;
    int wv   = threadIdx.x >> 5;
    int bh = blockIdx.y;
    int b = bh >> 4, h = bh & 15;
    int i0 = blockIdx.x * 128 + wv * 16;
    const bf16* qh = qb + (size_t)bh * SEQ * DIM_HEAD;
    const bf16* kh = kb + (size_t)bh * SEQ * DIM_HEAD;
    const bf16* vh = vT + (size_t)bh * DIM_HEAD * SEQ;
    int half = lane >> 4;
    bf16* sP = smem + 8192 + wv * 512;

    FragAB qa[2];
    {
        const bf16* qr = qh + (size_t)(i0 + (lane & 15)) * DIM_HEAD + ((lane >> 4) << 3);
#pragma unroll
        for (int kc = 0; kc < 2; kc++) {
            qa[kc].q[0] = *(const u32x4*)(qr + kc * 32);
            qa[kc].q[1] = *(const u32x4*)(qr + kc * 32 + 16);
        }
    }

    float mrow[8], lrow[8], alpha[8];
#pragma unroll
    for (int r = 0; r < 8; r++) { mrow[r] = -1e30f; lrow[r] = 0.0f; }
    v8f oacc[4] = {};
    const float sc = 0.125f;                        // 1/sqrt(64)

    if (wv == 0) {
        tdm_load_2d(smem,        0,    kh, 64, 32, DIM_HEAD, DIM_HEAD, SEQ); // K [32 s][64 d]
        tdm_load_2d(smem + 4096, 8192, vh, 32, 64, SEQ, SEQ, DIM_HEAD);      // V [64 n][32 s]
        __builtin_amdgcn_s_wait_tensorcnt(0);
    }
    __syncthreads();

    for (int j0 = 0; j0 < SEQ; j0 += 32) {
        int cur = (j0 >> 5) & 1;
        if (wv == 0 && (j0 + 32) < SEQ) {
            tdm_load_2d(smem + (cur ? 0 : 2048), cur ? 0 : 4096,
                        kh + (size_t)(j0 + 32) * DIM_HEAD, 64, 32, DIM_HEAD, DIM_HEAD, SEQ);
            tdm_load_2d(smem + (cur ? 4096 : 6144), cur ? 8192 : 12288,
                        vh + (j0 + 32), 32, 64, SEQ, SEQ, DIM_HEAD);
        }
        const bf16* sK = smem + cur * 2048;
        const bf16* sV = smem + 4096 + cur * 2048;

        // Batch all 4 K fragments, then 4 score WMMAs.
        FragAB kf[4];                                   // [jt*2 + kc]
#pragma unroll
        for (int jt = 0; jt < 2; jt++)
#pragma unroll
            for (int kc = 0; kc < 2; kc++) {
                const bf16* p = sK + (jt * 16 + (lane & 15)) * 64 + kc * 32 + ((lane >> 4) << 3);
                kf[jt * 2 + kc].q[0] = *(const u32x4*)p;
                kf[jt * 2 + kc].q[1] = *(const u32x4*)(p + 16);
            }
        v8f sf[2] = {};
        sf[0] = wmma_bf16(qa[0], kf[0], sf[0]);
        sf[0] = wmma_bf16(qa[1], kf[1], sf[0]);
        sf[1] = wmma_bf16(qa[0], kf[2], sf[1]);
        sf[1] = wmma_bf16(qa[1], kf[3], sf[1]);

        // Online softmax: lane owns rows r+8*half at column (lane&15)+16*jt.
#pragma unroll
        for (int r = 0; r < 8; r++) {
            float v0 = sf[0][r] * sc, v1 = sf[1][r] * sc;
            float mx = fmaxf(v0, v1);
#pragma unroll
            for (int msk = 8; msk >= 1; msk >>= 1) mx = fmaxf(mx, __shfl_xor(mx, msk, 32));
            float mnew = fmaxf(mrow[r], mx);
            float al = __expf(mrow[r] - mnew);
            float p0 = __expf(v0 - mnew), p1 = __expf(v1 - mnew);
            float rs = p0 + p1;
#pragma unroll
            for (int msk = 8; msk >= 1; msk >>= 1) rs += __shfl_xor(rs, msk, 32);
            lrow[r] = lrow[r] * al + rs;
            mrow[r] = mnew;
            alpha[r] = al;
            int irow = r + 8 * half;
            sP[irow * 32 + (lane & 15)]      = (bf16)p0;
            sP[irow * 32 + 16 + (lane & 15)] = (bf16)p1;
        }
#pragma unroll
        for (int t = 0; t < 4; t++)
#pragma unroll
            for (int r = 0; r < 8; r++) oacc[t][r] = oacc[t][r] * alpha[r];

        asm volatile("s_wait_dscnt 0x0" ::: "memory");  // wave-private LDS RAW

        // Batch P reload + all 4 V fragments, then 4 PV WMMAs.
        FragAB pa, vf[4];
        {
            const bf16* p = sP + (lane & 15) * 32 + ((lane >> 4) << 3);
            pa.q[0] = *(const u32x4*)p;
            pa.q[1] = *(const u32x4*)(p + 16);
        }
#pragma unroll
        for (int t = 0; t < 4; t++) {
            const bf16* p = sV + (t * 16 + (lane & 15)) * 32 + ((lane >> 4) << 3);
            vf[t].q[0] = *(const u32x4*)p;
            vf[t].q[1] = *(const u32x4*)(p + 16);
        }
#pragma unroll
        for (int t = 0; t < 4; t++)
            oacc[t] = wmma_bf16(pa, vf[t], oacc[t]);

        if (wv == 0) __builtin_amdgcn_s_wait_tensorcnt(0);
        __syncthreads();
    }
#pragma unroll
    for (int r = 0; r < 8; r++) {
        float inv = 1.0f / lrow[r];
        int s = i0 + r + 8 * half;
        size_t base = ((size_t)b * SEQ + s) * DIM + h * DIM_HEAD;
#pragma unroll
        for (int t = 0; t < 4; t++)
            ob[base + t * 16 + (lane & 15)] = (bf16)(oacc[t][r] * inv);
    }
}

// ---------------- Kernel 5: output projection + bias (fp32 out) ----------------
__global__ void __launch_bounds__(256, 1)
out_proj(const bf16* __restrict__ ab, const bf16* __restrict__ wp,
         const float* __restrict__ bias, float* __restrict__ out) {
    __shared__ __align__(16) bf16 smem[12288];      // 24 KB
    int lane = threadIdx.x & 31;
    int wv   = threadIdx.x >> 5;
    int wm = wv & 3, wn = wv >> 2;
    int m0 = blockIdx.y * 64 + wm * 16;
    int n0 = blockIdx.x * 128 + wn * 64;
    v8f acc[4] = {};
    gemm_mainloop(ab, wp, blockIdx.y * 64, blockIdx.x * 128, DIM, ROWS,
                  smem, lane, wv, wm, wn, acc);
    int half = lane >> 4;
#pragma unroll
    for (int t = 0; t < 4; t++) {
        int c = n0 + t * 16 + (lane & 15);
        float bv = bias[c];
#pragma unroll
        for (int r = 0; r < 8; r++) {
            int gr = m0 + r + 8 * half;
            out[(size_t)gr * DIM + c] = acc[t][r] + bv;
        }
    }
}

extern "C" void kernel_launch(void* const* d_in, const int* in_sizes, int n_in,
                              void* d_out, int out_size, void* d_ws, size_t ws_size,
                              hipStream_t stream) {
    (void)in_sizes; (void)n_in; (void)out_size; (void)ws_size;
    const float* x      = (const float*)d_in[0];
    const float* w_qkv  = (const float*)d_in[1];
    const float* w_out  = (const float*)d_in[2];
    const float* b_out  = (const float*)d_in[3];
    const float* gamma  = (const float*)d_in[4];
    const float* beta   = (const float*)d_in[5];
    float* out = (float*)d_out;

    char* ws = (char*)d_ws;
    size_t off = 0;
    bf16* xn    = (bf16*)(ws + off); off += (size_t)ROWS * DIM * 2;                    // 16 MB
    bf16* wqkvp = (bf16*)(ws + off); off += (size_t)QKV_N * DIM * 2;                   //  6 MB
    bf16* woutp = (bf16*)(ws + off); off += (size_t)DIM * DIM * 2;                     //  2 MB
    bf16* qb    = (bf16*)(ws + off); off += (size_t)BATCH * HEADS * SEQ * DIM_HEAD * 2;// 16 MB
    bf16* kb    = (bf16*)(ws + off); off += (size_t)BATCH * HEADS * SEQ * DIM_HEAD * 2;// 16 MB
    bf16* vT    = (bf16*)(ws + off); off += (size_t)BATCH * HEADS * SEQ * DIM_HEAD * 2;// 16 MB
    bf16* ob    = (bf16*)(ws + off); off += (size_t)ROWS * DIM * 2;                    // 16 MB

    pack_w_kernel<<<(QKV_N * DIM) / 256, 256, 0, stream>>>(w_qkv, wqkvp, DIM, QKV_N);
    pack_w_kernel<<<(DIM * DIM) / 256, 256, 0, stream>>>(w_out, woutp, DIM, DIM);
    ln_kernel<<<ROWS, 256, 0, stream>>>(x, gamma, beta, xn);
    qkv_gemm_rope<<<dim3(QKV_N / 128, ROWS / 64), 256, 0, stream>>>(xn, wqkvp, qb, kb, vT);
    attn_kernel<<<dim3(SEQ / 128, BATCH * HEADS), 256, 0, stream>>>(qb, kb, vT, ob);
    out_proj<<<dim3(DIM / 128, ROWS / 64), 256, 0, stream>>>(ob, woutp, b_out, out);
}
